// NodeAggregator_9672266350629
// MI455X (gfx1250) — matrix operand order repaired and checked by reading, not compiled, not run on hardware
//
#include <hip/hip_runtime.h>
#include <hip/hip_bf16.h>

// ---------------------------------------------------------------------------
// Problem constants (from reference)
// ---------------------------------------------------------------------------
#define NN      100000
#define DD      256
#define HH      256
#define BB      128
#define EE      128
#define STEPS   5
#define G4H     1024     // 4*H
#define H2      512      // 2*H

// ---------------------------------------------------------------------------
// CDNA5 WMMA types/helpers (wave32, 16x16x32 bf16)
// ---------------------------------------------------------------------------
typedef __bf16 bf16_t;
typedef bf16_t v16bf_t __attribute__((ext_vector_type(16)));
typedef float  v8f_t   __attribute__((ext_vector_type(8)));

__device__ __forceinline__ v8f_t wmma_bf16(v16bf_t a, v16bf_t b, v8f_t c) {
    return __builtin_amdgcn_wmma_f32_16x16x32_bf16(
        /*neg_a=*/false, a, /*neg_b=*/false, b,
        /*c_mod=*/(short)0, c, /*reuse_a=*/false, /*reuse_b=*/false);
}

__device__ __forceinline__ unsigned short f2bf_bits(float f) {
    bf16_t h = (bf16_t)f;
    return __builtin_bit_cast(unsigned short, h);
}

// Build 16x32 bf16 A-fragment from a row of f32:
//   lane l holds row M = l&15; koff = (l>=16)?8:0
//   elements 0..7  <- K = kbase+koff+0..7
//   elements 8..15 <- K = kbase+koff+16..23
__device__ __forceinline__ v16bf_t load_a_f32(const float* __restrict__ p) {
    v16bf_t a;
#pragma unroll
    for (int j = 0; j < 8; ++j) {
        a[j]     = (bf16_t)p[j];
        a[8 + j] = (bf16_t)p[16 + j];
    }
    return a;
}

// Load a pre-swizzled B-fragment (16 bf16 = 32B contiguous per lane)
__device__ __forceinline__ v16bf_t load_b_swz(const unsigned short* __restrict__ base,
                                              int kt, int nt, int ntilesN, int lane) {
    const unsigned short* p = base + ((size_t)(kt * ntilesN + nt) * 32 + lane) * 16;
    v16bf_t b;
    *(uint4*)&b       = *(const uint4*)p;
    *((uint4*)&b + 1) = *(const uint4*)(p + 8);
    return b;
}

__device__ __forceinline__ float sigmoidf_(float x) { return 1.f / (1.f + __expf(-x)); }

// ---------------------------------------------------------------------------
// Weight convert + swizzle into WMMA B-fragment order.
// dst layout: ((kt*ntilesN + nt)*32 + lane)*16 + j, where
//   n = nt*16 + (lane&15);  k = kt*32 + ((lane>=16)?8:0) + (j&7) + (j>>3)*16   (j = 0..15)
// transpose=0: B[k][n] = src[k*Nn + n];  transpose=1: B[k][n] = src[n*K + k]
// ---------------------------------------------------------------------------
__global__ void k_cvt_swz(const float* __restrict__ src, unsigned short* __restrict__ dst,
                          int K, int Nn, int transpose) {
    int total = K * Nn;
    int ntilesN = Nn >> 4;
    for (int idx = blockIdx.x * blockDim.x + threadIdx.x; idx < total;
         idx += gridDim.x * blockDim.x) {
        int j    = idx & 15;
        int l    = (idx >> 4) & 31;
        int tile = idx >> 9;
        int nt   = tile % ntilesN;
        int kt   = tile / ntilesN;
        int n    = nt * 16 + (l & 15);
        int k    = kt * 32 + ((l >> 4) & 1) * 8 + (j & 7) + (j >> 3) * 16;
        float v  = transpose ? src[(size_t)n * K + k] : src[(size_t)k * Nn + n];
        dst[idx] = f2bf_bits(v);
    }
}

__global__ void k_fill(float* __restrict__ p, int n, float v) {
    for (int i = blockIdx.x * blockDim.x + threadIdx.x; i < n; i += gridDim.x * blockDim.x)
        p[i] = v;
}

// ---------------------------------------------------------------------------
// FNN: hfeat = elu(x@W1 + b1)@W2 + b2, fused, W1+W2 bf16 resident in LDS.
// block = 128 threads (4 waves); each wave computes a 16x256 row strip.
// Weight staging uses CDNA5 async-to-LDS (ASYNCcnt) — no VGPR round-trip.
// A-fragments for the whole K range are cached in registers (8 x v16bf = 64
// VGPRs); FOUR independent accumulator chains hide the WMMA->WMMA RAW hazard
// (bf16 WMMA needs ~5 slots between dependent ops, ISA 7.12.1).
// Dynamic LDS: W1 (128KB) + W2 (128KB) + 4x 16x256 bf16 scratch (32KB) = 288KB
// ---------------------------------------------------------------------------
#define FNN_WAVES 4
#define FNN_SMEM  (131072 + 131072 + FNN_WAVES * 16 * 256 * 2)

__global__ void k_fnn(const float* __restrict__ x,
                      const float* __restrict__ b1v, const float* __restrict__ b2v,
                      const unsigned short* __restrict__ W1sw,
                      const unsigned short* __restrict__ W2sw,
                      float* __restrict__ hfeat, int nrows) {
    extern __shared__ char smem[];
    unsigned short* sW1  = (unsigned short*)smem;          // 65536 bf16
    unsigned short* sW2  = sW1 + 65536;                    // 65536 bf16
    unsigned short* sScr = sW2 + 65536;                    // per-wave 16x256 bf16

    // ---- async weight staging: 256KB global -> LDS via ASYNCcnt path ----
    {
        unsigned lds0 = (unsigned)(size_t)smem;            // LDS offset = addr[31:0]
        unsigned long long g1 = (unsigned long long)W1sw;
        unsigned long long g2 = (unsigned long long)W2sw;
        for (int i = threadIdx.x * 16; i < 131072; i += blockDim.x * 16) {
            unsigned d1 = lds0 + i;
            unsigned long long s1 = g1 + i;
            asm volatile("global_load_async_to_lds_b128 %0, %1, off"
                         :: "v"(d1), "v"(s1) : "memory");
            unsigned d2 = lds0 + 131072 + i;
            unsigned long long s2 = g2 + i;
            asm volatile("global_load_async_to_lds_b128 %0, %1, off"
                         :: "v"(d2), "v"(s2) : "memory");
        }
        asm volatile("s_wait_asynccnt 0x0" ::: "memory");
    }
    __syncthreads();

    const int lane = threadIdx.x & 31;
    const int wave = threadIdx.x >> 5;
    unsigned short* scr = sScr + wave * (16 * 256);

    const int gwave  = blockIdx.x * FNN_WAVES + wave;
    const int nwaves = gridDim.x * FNN_WAVES;
    const int ntiles = nrows >> 4;            // 6250 row tiles (16 | 100000)

    const int arow_l = lane & 15;
    const int akoff  = ((lane >> 4) & 1) * 8;
    const int cn     = lane & 15;
    const int cm0    = ((lane >> 4) & 1) * 8;

    const v8f_t zero8 = {0.f, 0.f, 0.f, 0.f, 0.f, 0.f, 0.f, 0.f};

    for (int tile = gwave; tile < ntiles; tile += nwaves) {
        const int row0 = tile * 16;

        // ---- cache GEMM1 A-fragments: read the x row tile exactly once ----
        v16bf_t a1[8];
#pragma unroll
        for (int kt = 0; kt < 8; ++kt)
            a1[kt] = load_a_f32(x + (size_t)(row0 + arow_l) * DD + kt * 32 + akoff);

        // ---- GEMM1 + bias + ELU -> bf16 scratch; 4 acc chains for ILP ----
#pragma unroll 1
        for (int nt = 0; nt < 16; nt += 4) {
            v8f_t acc0 = zero8, acc1 = zero8, acc2 = zero8, acc3 = zero8;
#pragma unroll
            for (int kt = 0; kt < 8; ++kt) {
                acc0 = wmma_bf16(a1[kt], load_b_swz(sW1, kt, nt,     16, lane), acc0);
                acc1 = wmma_bf16(a1[kt], load_b_swz(sW1, kt, nt + 1, 16, lane), acc1);
                acc2 = wmma_bf16(a1[kt], load_b_swz(sW1, kt, nt + 2, 16, lane), acc2);
                acc3 = wmma_bf16(a1[kt], load_b_swz(sW1, kt, nt + 3, 16, lane), acc3);
            }
#pragma unroll
            for (int q = 0; q < 4; ++q) {
                v8f_t av = (q == 0) ? acc0 : (q == 1) ? acc1 : (q == 2) ? acc2 : acc3;
                float bias = b1v[(nt + q) * 16 + cn];
#pragma unroll
                for (int r = 0; r < 8; ++r) {
                    float v = av[r] + bias;
                    v = (v > 0.f) ? v : (__expf(v) - 1.f);   // ELU(alpha=1)
                    scr[(cm0 + r) * 256 + (nt + q) * 16 + cn] = f2bf_bits(v);
                }
            }
        }
        asm volatile("s_wait_dscnt 0x0" ::: "memory");       // LDS RAW handoff

        // ---- cache GEMM2 A-fragments from scratch (ds_load_b128 x2 each) ----
        v16bf_t a2[8];
#pragma unroll
        for (int kt = 0; kt < 8; ++kt) {
            const unsigned short* sp = scr + arow_l * 256 + kt * 32 + akoff;
            v16bf_t a;
            *(uint4*)&a       = *(const uint4*)sp;           // K = kbase+0..7
            *((uint4*)&a + 1) = *(const uint4*)(sp + 16);    // K = kbase+16..23
            a2[kt] = a;
        }

        // ---- GEMM2 + bias b2 -> hfeat (f32) ----
#pragma unroll 1
        for (int nt = 0; nt < 16; nt += 4) {
            v8f_t acc0 = zero8, acc1 = zero8, acc2 = zero8, acc3 = zero8;
#pragma unroll
            for (int kt = 0; kt < 8; ++kt) {
                acc0 = wmma_bf16(a2[kt], load_b_swz(sW2, kt, nt,     16, lane), acc0);
                acc1 = wmma_bf16(a2[kt], load_b_swz(sW2, kt, nt + 1, 16, lane), acc1);
                acc2 = wmma_bf16(a2[kt], load_b_swz(sW2, kt, nt + 2, 16, lane), acc2);
                acc3 = wmma_bf16(a2[kt], load_b_swz(sW2, kt, nt + 3, 16, lane), acc3);
            }
#pragma unroll
            for (int q = 0; q < 4; ++q) {
                v8f_t av = (q == 0) ? acc0 : (q == 1) ? acc1 : (q == 2) ? acc2 : acc3;
                float bias = b2v[(nt + q) * 16 + cn];
#pragma unroll
                for (int r = 0; r < 8; ++r) {
                    hfeat[(size_t)(row0 + cm0 + r) * HH + (nt + q) * 16 + cn] =
                        av[r] + bias;
                }
            }
        }
    }
}

// ---------------------------------------------------------------------------
// LSTM gates: gates = xin@Wih^T + h@Whh^T + bih + bhh   (B x 4H)
// one wave per 16x16 output tile: mt in [0,8), nt in [0,64).
// Two partial-sum chains (even/odd kt) halve the dependent-WMMA chain depth.
// ---------------------------------------------------------------------------
__global__ void k_gates(const float* __restrict__ xin, int Kin,
                        const float* __restrict__ h,
                        const unsigned short* __restrict__ WihSw,
                        const unsigned short* __restrict__ WhhSw,
                        const float* __restrict__ bih, const float* __restrict__ bhh,
                        float* __restrict__ gates) {
    const int gw   = (blockIdx.x * blockDim.x + threadIdx.x) >> 5;
    const int lane = threadIdx.x & 31;
    const int mt = gw >> 6;
    const int nt = gw & 63;
    if (mt >= 8) return;

    const int arow  = mt * 16 + (lane & 15);
    const int akoff = ((lane >> 4) & 1) * 8;
    const v8f_t zero8 = {0.f, 0.f, 0.f, 0.f, 0.f, 0.f, 0.f, 0.f};
    v8f_t acc0 = zero8, acc1 = zero8;

#pragma unroll 1
    for (int kt = 0; kt < (Kin >> 5); kt += 2) {     // Kin/32 is 16 or 8 (even)
        v16bf_t ae = load_a_f32(xin + (size_t)arow * Kin + kt * 32 + akoff);
        v16bf_t ao = load_a_f32(xin + (size_t)arow * Kin + (kt + 1) * 32 + akoff);
        acc0 = wmma_bf16(ae, load_b_swz(WihSw, kt,     nt, 64, lane), acc0);
        acc1 = wmma_bf16(ao, load_b_swz(WihSw, kt + 1, nt, 64, lane), acc1);
    }
#pragma unroll 1
    for (int kt = 0; kt < 8; kt += 2) {              // H/32 = 8
        v16bf_t ae = load_a_f32(h + (size_t)arow * HH + kt * 32 + akoff);
        v16bf_t ao = load_a_f32(h + (size_t)arow * HH + (kt + 1) * 32 + akoff);
        acc0 = wmma_bf16(ae, load_b_swz(WhhSw, kt,     nt, 64, lane), acc0);
        acc1 = wmma_bf16(ao, load_b_swz(WhhSw, kt + 1, nt, 64, lane), acc1);
    }

    const int cn  = nt * 16 + (lane & 15);
    const int cm0 = ((lane >> 4) & 1) * 8;
    const float bias = bih[cn] + bhh[cn];
#pragma unroll
    for (int r = 0; r < 8; ++r)
        gates[(size_t)(mt * 16 + cm0 + r) * G4H + cn] = acc0[r] + acc1[r] + bias;
}

// LSTM elementwise cell update (PyTorch gate order i,f,g,o)
__global__ void k_cell(const float* __restrict__ gates,
                       float* __restrict__ c, float* __restrict__ h) {
    int idx = blockIdx.x * blockDim.x + threadIdx.x;
    if (idx >= BB * HH) return;
    int b = idx >> 8, j = idx & 255;
    const float* g = gates + (size_t)b * G4H;
    float ig = sigmoidf_(g[j]);
    float fg = sigmoidf_(g[256 + j]);
    float gg = tanhf(g[512 + j]);
    float og = sigmoidf_(g[768 + j]);
    float cc = fg * c[idx] + ig * gg;
    c[idx] = cc;
    h[idx] = og * tanhf(cc);
}

// ---------------------------------------------------------------------------
// Attention (per step).  hfeat (102MB) is L2-resident on MI455X (192MB L2).
// ---------------------------------------------------------------------------
__global__ void k_step_reset(float* __restrict__ m, float* __restrict__ denom,
                             float* __restrict__ r) {
    int i = blockIdx.x * blockDim.x + threadIdx.x;
    if (i < BB) { m[i] = -3.402823466e38f; denom[i] = 0.f; }
    if (i < BB * HH) r[i] = 0.f;
}

__device__ __forceinline__ void atomicMaxF(float* addr, float val) {
    unsigned int* u = (unsigned int*)addr;
    unsigned int old = *u;
    while (__uint_as_float(old) < val) {
        unsigned int assumed = old;
        old = atomicCAS(u, assumed, __float_as_uint(val));
        if (old == assumed) break;
    }
}

// e[n] = <hfeat[n,:], q[idx[n],:]>, m[b] = segment max.  One wave per node.
__global__ void k_attn_e(const float* __restrict__ hfeat, const float* __restrict__ q,
                         const int* __restrict__ idx, float* __restrict__ e,
                         float* __restrict__ m, int n_nodes) {
    int gw   = (blockIdx.x * blockDim.x + threadIdx.x) >> 5;
    int lane = threadIdx.x & 31;
    if (gw >= n_nodes) return;
    int b = idx[gw];
    const float* hp = hfeat + (size_t)gw * HH + lane * 8;
    const float* qp = q + (size_t)b * HH + lane * 8;
    float s = 0.f;
#pragma unroll
    for (int j = 0; j < 8; ++j) s += hp[j] * qp[j];
#pragma unroll
    for (int off = 16; off > 0; off >>= 1) s += __shfl_xor(s, off, 32);
    if (lane == 0) {
        e[gw] = s;
        atomicMaxF(&m[b], s);
    }
}

// ex[n] = exp(e[n]-m[idx]); denom[b] += ex
__global__ void k_attn_s(const float* __restrict__ e, const int* __restrict__ idx,
                         const float* __restrict__ m, float* __restrict__ ex,
                         float* __restrict__ denom, int n_nodes) {
    int n = blockIdx.x * blockDim.x + threadIdx.x;
    if (n >= n_nodes) return;
    int b = idx[n];
    float mb = m[b];
    if (!(mb > -1e30f)) mb = 0.f;                 // empty-segment guard
    float v = __expf(e[n] - mb);
    ex[n] = v;
    atomicAdd(&denom[b], v);
}

// r[b,:] += a_n * hfeat[n,:].  batch_idxs sorted -> register accumulation
// per segment run, atomic flush only at boundaries. thread t = feature col.
#define ATTN_CHUNK 128
__global__ void k_attn_r(const float* __restrict__ hfeat, const float* __restrict__ ex,
                         const int* __restrict__ idx, const float* __restrict__ denom,
                         float* __restrict__ r, int n_nodes) {
    int start = blockIdx.x * ATTN_CHUNK;
    if (start >= n_nodes) return;
    int end = start + ATTN_CHUNK;
    if (end > n_nodes) end = n_nodes;
    int t = threadIdx.x;                          // 0..255 feature column

    int cur = idx[start];
    float acc = 0.f;
    for (int n = start; n < end; ++n) {
        __builtin_prefetch(hfeat + (size_t)(n + 4) * HH + t, 0, 1);
        int bi = idx[n];
        if (bi != cur) {
            atomicAdd(&r[(size_t)cur * HH + t], acc);
            acc = 0.f;
            cur = bi;
        }
        float a = ex[n] / (denom[bi] + 1e-16f);
        acc += a * hfeat[(size_t)n * HH + t];
    }
    atomicAdd(&r[(size_t)cur * HH + t], acc);
}

// q_star = concat(h2, r)
__global__ void k_qstar(const float* __restrict__ h2, const float* __restrict__ r,
                        float* __restrict__ q_star) {
    int i = blockIdx.x * blockDim.x + threadIdx.x;
    if (i >= BB * H2) return;
    int b = i >> 9, j = i & 511;
    q_star[i] = (j < 256) ? h2[(size_t)b * HH + j] : r[(size_t)b * HH + (j - 256)];
}

// out = q_star @ out_W + out_b : (128x512)@(512x128); one wave per 16x16 tile
__global__ void k_out(const float* __restrict__ q_star,
                      const unsigned short* __restrict__ outWsw,
                      const float* __restrict__ out_b, float* __restrict__ out) {
    const int gw   = (blockIdx.x * blockDim.x + threadIdx.x) >> 5;
    const int lane = threadIdx.x & 31;
    const int mt = gw >> 3;
    const int nt = gw & 7;
    if (mt >= 8) return;

    const int arow  = mt * 16 + (lane & 15);
    const int akoff = ((lane >> 4) & 1) * 8;
    const v8f_t zero8 = {0.f, 0.f, 0.f, 0.f, 0.f, 0.f, 0.f, 0.f};
    v8f_t acc0 = zero8, acc1 = zero8;
#pragma unroll 1
    for (int kt = 0; kt < 16; kt += 2) {          // K = 512
        v16bf_t ae = load_a_f32(q_star + (size_t)arow * H2 + kt * 32 + akoff);
        v16bf_t ao = load_a_f32(q_star + (size_t)arow * H2 + (kt + 1) * 32 + akoff);
        acc0 = wmma_bf16(ae, load_b_swz(outWsw, kt,     nt, 8, lane), acc0);
        acc1 = wmma_bf16(ao, load_b_swz(outWsw, kt + 1, nt, 8, lane), acc1);
    }
    const int cn  = nt * 16 + (lane & 15);
    const int cm0 = ((lane >> 4) & 1) * 8;
    const float bias = out_b[cn];
#pragma unroll
    for (int r = 0; r < 8; ++r)
        out[(size_t)(mt * 16 + cm0 + r) * EE + cn] = acc0[r] + acc1[r] + bias;
}

// ---------------------------------------------------------------------------
// Host side
// ---------------------------------------------------------------------------
extern "C" void kernel_launch(void* const* d_in, const int* in_sizes, int n_in,
                              void* d_out, int out_size, void* d_ws, size_t ws_size,
                              hipStream_t stream) {
    (void)in_sizes; (void)n_in; (void)out_size; (void)ws_size;

    const float* x      = (const float*)d_in[0];
    const int*   bidx   = (const int*)  d_in[1];
    const float* fnn_W1 = (const float*)d_in[2];
    const float* fnn_b1 = (const float*)d_in[3];
    const float* fnn_W2 = (const float*)d_in[4];
    const float* fnn_b2 = (const float*)d_in[5];
    const float* Wih0   = (const float*)d_in[6];
    const float* Whh0   = (const float*)d_in[7];
    const float* bih0   = (const float*)d_in[8];
    const float* bhh0   = (const float*)d_in[9];
    const float* Wih1   = (const float*)d_in[10];
    const float* Whh1   = (const float*)d_in[11];
    const float* bih1   = (const float*)d_in[12];
    const float* bhh1   = (const float*)d_in[13];
    const float* Wih2   = (const float*)d_in[14];
    const float* Whh2   = (const float*)d_in[15];
    const float* bih2   = (const float*)d_in[16];
    const float* bhh2   = (const float*)d_in[17];
    const float* out_W  = (const float*)d_in[18];
    const float* out_b  = (const float*)d_in[19];
    float* out = (float*)d_out;

    // ---- workspace partition ----
    char* base = (char*)d_ws;
    size_t off = 0;
    auto alloc = [&](size_t bytes) -> char* {
        char* p = base + off;
        off = (off + bytes + 255) & ~(size_t)255;
        return p;
    };
    float* hfeat = (float*)alloc((size_t)NN * HH * 4);           // 102.4 MB
    unsigned short* W1sw   = (unsigned short*)alloc((size_t)DD * HH * 2);
    unsigned short* W2sw   = (unsigned short*)alloc((size_t)HH * HH * 2);
    unsigned short* Wih0sw = (unsigned short*)alloc((size_t)H2 * G4H * 2);
    unsigned short* Whh0sw = (unsigned short*)alloc((size_t)HH * G4H * 2);
    unsigned short* Wih1sw = (unsigned short*)alloc((size_t)HH * G4H * 2);
    unsigned short* Whh1sw = (unsigned short*)alloc((size_t)HH * G4H * 2);
    unsigned short* Wih2sw = (unsigned short*)alloc((size_t)HH * G4H * 2);
    unsigned short* Whh2sw = (unsigned short*)alloc((size_t)HH * G4H * 2);
    unsigned short* outWsw = (unsigned short*)alloc((size_t)H2 * EE * 2);
    float* state  = (float*)alloc((size_t)(6 * BB * HH + BB * H2) * 4); // h0..c2,q_star
    float* h0 = state;
    float* h1 = h0 + BB * HH;
    float* h2 = h1 + BB * HH;
    float* c0 = h2 + BB * HH;
    float* c1 = c0 + BB * HH;
    float* c2 = c1 + BB * HH;
    float* q_star = c2 + BB * HH;
    float* gates = (float*)alloc((size_t)BB * G4H * 4);
    float* e     = (float*)alloc((size_t)NN * 4);
    float* ex    = (float*)alloc((size_t)NN * 4);
    float* m     = (float*)alloc((size_t)BB * 4);
    float* denom = (float*)alloc((size_t)BB * 4);
    float* r     = (float*)alloc((size_t)BB * HH * 4);

    // ---- weight convert + swizzle into WMMA B-fragment order ----
    auto cvt = [&](const float* src, unsigned short* dst, int K, int Nv, int tr) {
        int total = K * Nv;
        k_cvt_swz<<<(total + 255) / 256, 256, 0, stream>>>(src, dst, K, Nv, tr);
    };
    cvt(fnn_W1, W1sw,   DD, HH,  0);
    cvt(fnn_W2, W2sw,   HH, HH,  0);
    cvt(Wih0,   Wih0sw, H2, G4H, 1);
    cvt(Whh0,   Whh0sw, HH, G4H, 1);
    cvt(Wih1,   Wih1sw, HH, G4H, 1);
    cvt(Whh1,   Whh1sw, HH, G4H, 1);
    cvt(Wih2,   Wih2sw, HH, G4H, 1);
    cvt(Whh2,   Whh2sw, HH, G4H, 1);
    cvt(out_W,  outWsw, H2, EE,  0);

    // ---- zero LSTM state + q_star ----
    {
        int n = 6 * BB * HH + BB * H2;
        k_fill<<<(n + 255) / 256, 256, 0, stream>>>(state, n, 0.f);
    }

    // ---- FNN (fused 2 GEMMs, 256KB of weights in LDS) ----
    k_fnn<<<512, 128, FNN_SMEM, stream>>>(x, fnn_b1, fnn_b2, W1sw, W2sw, hfeat, NN);

    // ---- STEPS x (3-layer LSTM + segment attention) ----
    for (int s = 0; s < STEPS; ++s) {
        k_gates<<<64, 256, 0, stream>>>(q_star, H2, h0, Wih0sw, Whh0sw, bih0, bhh0, gates);
        k_cell<<<(BB * HH + 255) / 256, 256, 0, stream>>>(gates, c0, h0);
        k_gates<<<64, 256, 0, stream>>>(h0, HH, h1, Wih1sw, Whh1sw, bih1, bhh1, gates);
        k_cell<<<(BB * HH + 255) / 256, 256, 0, stream>>>(gates, c1, h1);
        k_gates<<<64, 256, 0, stream>>>(h1, HH, h2, Wih2sw, Whh2sw, bih2, bhh2, gates);
        k_cell<<<(BB * HH + 255) / 256, 256, 0, stream>>>(gates, c2, h2);

        k_step_reset<<<(BB * HH + 255) / 256, 256, 0, stream>>>(m, denom, r);
        k_attn_e<<<(NN * 32 + 255) / 256, 256, 0, stream>>>(hfeat, h2, bidx, e, m, NN);
        k_attn_s<<<(NN + 255) / 256, 256, 0, stream>>>(e, bidx, m, ex, denom, NN);
        k_attn_r<<<(NN + ATTN_CHUNK - 1) / ATTN_CHUNK, 256, 0, stream>>>(hfeat, ex, bidx,
                                                                         denom, r, NN);
        k_qstar<<<(BB * H2 + 255) / 256, 256, 0, stream>>>(h2, r, q_star);
    }

    // ---- output head ----
    k_out<<<8, 256, 0, stream>>>(q_star, outWsw, out_b, out);
}
